// MySDPA_77592879170154
// MI455X (gfx1250) — matrix operand order repaired and checked by reading, not compiled, not run on hardware
//
#include <hip/hip_runtime.h>
#include <hip/hip_bf16.h>
#include <math.h>

typedef __attribute__((ext_vector_type(16))) __bf16 v16bf;
typedef __attribute__((ext_vector_type(8)))  __bf16 v8bf;
typedef __attribute__((ext_vector_type(8)))  float  v8f;

// Problem constants (match reference)
constexpr int Bc = 2, Hc = 16, Sc = 2048, Dc = 128;
constexpr int BLOCK_M = 128;  // 8 waves x 16 query rows
constexpr int BLOCK_N = 64;   // keys per tile (amortizes softmax fixed costs)
constexpr int NSUB    = BLOCK_N / 16;  // 4 key sub-tiles
constexpr int NWAVES  = 8;
constexpr int DVT     = Dc + 16;       // V^T rows: 128 d-rows + 16 rows of 1.0
                                       // (ones-tile turns row-sum into a WMMA)

__device__ __forceinline__ float rowmax16(float v) {
    v = fmaxf(v, __shfl_xor(v, 1, 32));
    v = fmaxf(v, __shfl_xor(v, 2, 32));
    v = fmaxf(v, __shfl_xor(v, 4, 32));
    v = fmaxf(v, __shfl_xor(v, 8, 32));
    return v;
}

__global__ __launch_bounds__(256)
void sdpa_flash_wmma(const float* __restrict__ Q,
                     const float* __restrict__ K,
                     const float* __restrict__ V,
                     float* __restrict__ O)
{
    // LDS: K tile row-major [key][d], V tile transposed [d][key] (+ones rows),
    // P scratch per wave.
    __shared__ __bf16 Kl[BLOCK_N * Dc];            // 16 KB
    __shared__ __bf16 Vt[DVT * BLOCK_N];           // 18 KB
    __shared__ __bf16 Pl[NWAVES * 16 * BLOCK_N];   // 16 KB

    const int tid  = threadIdx.x;
    const int wave = tid >> 5;
    const int lane = tid & 31;
    const int half = lane >> 4;   // 0: lanes 0-15, 1: lanes 16-31
    const int l16  = lane & 15;

    const int bh = blockIdx.y;                 // fused (b, h)
    const int qb = blockIdx.x * BLOCK_M;       // first query row of this block
    const size_t base = (size_t)bh * Sc * Dc;

    // 1/sqrt(128) * log2(e): softmax in the exp2 domain; scale folded into the
    // Q->bf16 conversion (mask uses -inf, so mask/scale order is equivalent).
    const float qscale = 0.08838834764831845f * 1.4426950408889634f;

    // ---- One-time init: ones-rows of V^T (rows Dc..Dc+15) ----
#pragma unroll
    for (int e = 0; e < (16 * BLOCK_N) / 256; ++e)
        Vt[Dc * BLOCK_N + e * 256 + tid] = (__bf16)1.0f;
    // (visible to all waves after the tile-load barrier in iteration 0)

    // ---- Load this wave's Q stripe as WMMA A-fragments (fp32 -> bf16) ----
    // A 16x32, contiguous k order used here (B matches): lane holds row M=l16; VGPR0-3: K = 16*half+0..7,
    // VGPR4-7: K = 16*half+8..15  (per K-chunk of 32)
    const int qrow_a = qb + wave * 16 + l16;
    v16bf qa[4];
    {
        const float* qrow = Q + base + (size_t)qrow_a * Dc;
#pragma unroll
        for (int c = 0; c < 4; ++c) {
            v8f lo = *(const v8f*)(qrow + 32 * c + 16 * half);
            v8f hi = *(const v8f*)(qrow + 32 * c + 16 * half + 8);
#pragma unroll
            for (int e = 0; e < 8; ++e) {
                qa[c][e]     = (__bf16)(lo[e] * qscale);
                qa[c][e + 8] = (__bf16)(hi[e] * qscale);
            }
        }
    }

    // ---- Flash-attention state ----
    // acc[0..7]: 16x128 f32 output accumulator; acc[8]: row-sum tile (P @ 1),
    // which makes the softmax denominator ride along with the accumulator
    // (rescaled by alpha together, no separate l update needed).
    v8f acc[9];
    const v8f vzero = {};
#pragma unroll
    for (int t = 0; t < 9; ++t) acc[t] = vzero;

    float m_r[8];               // per-row running max (row = r + 8*half)
#pragma unroll
    for (int r = 0; r < 8; ++r) m_r[r] = -INFINITY;

    const int row_lo  = qb + wave * 16;        // wave's first query row
    const int row_hi  = row_lo + 15;           // wave's last query row
    const int kmax    = qb + BLOCK_M;          // causal loop bound

    for (int kb = 0; kb < kmax; kb += BLOCK_N) {
        // ---- Prefetch next K/V tile into L2 (global_prefetch_b8) ----
        const int kbn = kb + BLOCK_N;
        if (kbn < kmax) {
#pragma unroll
            for (int i = 0; i < (BLOCK_N * Dc) / (256 * 8); ++i) {
                int idx8 = (i * 256 + tid) * 8;
                int key = idx8 >> 7, d = idx8 & 127;
                __builtin_prefetch(K + base + (size_t)(kbn + key) * Dc + d, 0, 1);
                __builtin_prefetch(V + base + (size_t)(kbn + key) * Dc + d, 0, 1);
            }
        }

        __syncthreads();  // previous iteration's LDS reads done

        // ---- Cooperative K/V tile load: v8f global loads, fp32 -> bf16 ----
#pragma unroll
        for (int i = 0; i < (BLOCK_N * Dc) / (256 * 8); ++i) {
            int idx8 = (i * 256 + tid) * 8;     // 8-element chunk index
            int key = idx8 >> 7, d = idx8 & 127;
            v8f kv = *(const v8f*)(K + base + (size_t)(kb + key) * Dc + d);
            v8f vv = *(const v8f*)(V + base + (size_t)(kb + key) * Dc + d);
            v8bf kp;
#pragma unroll
            for (int e = 0; e < 8; ++e) kp[e] = (__bf16)kv[e];
            *(v8bf*)&Kl[idx8] = kp;             // packed ds_store (16B aligned)
#pragma unroll
            for (int e = 0; e < 8; ++e)         // transpose scatter for V
                Vt[(d + e) * BLOCK_N + key] = (__bf16)vv[e];
        }
        __syncthreads();

        // Waves whose rows are entirely above this key tile contribute nothing.
        if (kb <= row_hi) {
            // ---- S = Q @ K^T : 4 key-subtiles x 4 K-chunks of WMMA ----
            v8f st[NSUB];
#pragma unroll
            for (int s = 0; s < NSUB; ++s) {
                v8f c = {};
                const int key = 16 * s + l16;   // B-frag: lane holds column N=key
#pragma unroll
                for (int cc = 0; cc < 4; ++cc) {
                    // B 32x16: lane's 16 K-values = d = 32*cc + 16*half + 0..15
                    const v16bf kbf =
                        *(const v16bf*)(&Kl[key * Dc + 32 * cc + 16 * half]);
                    c = __builtin_amdgcn_wmma_f32_16x16x32_bf16(
                            false, qa[cc], false, kbf, (short)0, c, false, false);
                }
                st[s] = c;
            }

            // ---- causal mask (diagonal tiles only; wave-uniform branch) ----
            if (kb + BLOCK_N - 1 > row_lo) {
#pragma unroll
                for (int r = 0; r < 8; ++r) {
                    const int row = row_lo + r + 8 * half;
#pragma unroll
                    for (int s = 0; s < NSUB; ++s) {
                        const int key = kb + 16 * s + l16;
                        st[s][r] = (key > row) ? -INFINITY : st[s][r];
                    }
                }
            }

            // ---- online softmax (exp2 domain) ----
            float alpha[8];
            int upd = 0;
#pragma unroll
            for (int r = 0; r < 8; ++r) {
                float rm = fmaxf(fmaxf(st[0][r], st[1][r]),
                                 fmaxf(st[2][r], st[3][r]));
                rm = rowmax16(rm);
                float m_new = fmaxf(m_r[r], rm);
                upd |= (m_new > m_r[r]);
                alpha[r] = exp2f(m_r[r] - m_new);
#pragma unroll
                for (int s = 0; s < NSUB; ++s)
                    st[s][r] = exp2f(st[s][r] - m_new);
                m_r[r] = m_new;
            }
            // Rescale only when some row's max advanced (wave-uniform check);
            // acc[8] (the P@1 row-sum tile) is rescaled along with the output.
            if (__any(upd)) {
#pragma unroll
                for (int t = 0; t < 9; ++t)
#pragma unroll
                    for (int r = 0; r < 8; ++r)
                        acc[t][r] *= alpha[r];
            }

            // ---- C-layout P -> A-layout via per-wave LDS scratch ----
            __bf16* pw = &Pl[wave * 16 * BLOCK_N];
#pragma unroll
            for (int s = 0; s < NSUB; ++s)
#pragma unroll
                for (int r = 0; r < 8; ++r)
                    pw[(r + 8 * half) * BLOCK_N + 16 * s + l16] =
                        (__bf16)st[s][r];

            // same-wave LDS RAW: wait on DS counter + fence compiler reordering
            asm volatile("s_wait_dscnt 0" ::: "memory");

            v16bf pa[2];  // A-fragments of P (16 x 64 -> two 16x32 chunks)
#pragma unroll
            for (int cp = 0; cp < 2; ++cp) {
                v8bf plo = *(const v8bf*)(pw + l16 * BLOCK_N + 32 * cp + 16 * half);
                v8bf phi = *(const v8bf*)(pw + l16 * BLOCK_N + 32 * cp + 16 * half + 8);
#pragma unroll
                for (int e = 0; e < 8; ++e) {
                    pa[cp][e] = plo[e];
                    pa[cp][e + 8] = phi[e];
                }
            }

            // ---- acc += P @ [V | 1] : 9 tiles x 2 key-chunks of WMMA ----
#pragma unroll
            for (int t = 0; t < 9; ++t) {
#pragma unroll
                for (int cp = 0; cp < 2; ++cp) {
                    // B 32x16: lane holds col N (= d for t<8, ones for t==8);
                    // K = keys 32*cp + 16*half + 0..15
                    const v16bf vbf = *(const v16bf*)(
                        &Vt[(16 * t + l16) * BLOCK_N + 32 * cp + 16 * half]);
                    acc[t] = __builtin_amdgcn_wmma_f32_16x16x32_bf16(
                                false, pa[cp], false, vbf, (short)0, acc[t],
                                false, false);
                }
            }
        }
    }

    // ---- normalize by row sum (acc[8]) and store ----
    // C layout: row = r + 8*half, col = 16t + l16; acc[8] holds the row sum in
    // every lane (all 16 ones-columns identical), so no broadcast is needed.
#pragma unroll
    for (int r = 0; r < 8; ++r) {
        const float inv = 1.0f / acc[8][r];
        const int row = row_lo + r + 8 * half;
        float* orow = O + base + (size_t)row * Dc;
#pragma unroll
        for (int t = 0; t < 8; ++t)
            orow[16 * t + l16] = acc[t][r] * inv;
    }
}

extern "C" void kernel_launch(void* const* d_in, const int* in_sizes, int n_in,
                              void* d_out, int out_size, void* d_ws, size_t ws_size,
                              hipStream_t stream) {
    const float* Q = (const float*)d_in[0];
    const float* K = (const float*)d_in[1];
    const float* V = (const float*)d_in[2];
    // d_in[3] is the causal mask; causality is computed analytically in-kernel.
    float* O = (float*)d_out;

    dim3 grid(Sc / BLOCK_M, Bc * Hc);
    dim3 block(256);
    sdpa_flash_wmma<<<grid, block, 0, stream>>>(Q, K, V, O);
}
